// SuperPointMagicLeap_13786845020167
// MI455X (gfx1250) — matrix operand order repaired and checked by reading, not compiled, not run on hardware
//
#include <hip/hip_runtime.h>
#include <hip/hip_bf16.h>
#include <math.h>
#include <stdint.h>

typedef __attribute__((ext_vector_type(16))) _Float16 v16h;
typedef __attribute__((ext_vector_type(8)))  _Float16 v8h;
typedef __attribute__((ext_vector_type(8)))  float    v8f;

#define BM 256     // pixels (GEMM M) per workgroup (1 pixel / thread for staging)
#define BN 64      // output channels (GEMM N) per workgroup
#define BK 32      // K-slice per WMMA step
#define AS_STR 40  // 32 + 8 halves pad (row stride, 80B -> 16B aligned chunks)

// ---------------------------------------------------------------- utilities
__global__ void convert_image_f32_to_f16(const float* __restrict__ in,
                                         _Float16* __restrict__ out, int n) {
  int i = blockIdx.x * blockDim.x + threadIdx.x;
  if (i < n) out[i] = (_Float16)in[i];
}

// OIHW f32 -> packed f16 [coutp][k] (fragment-major), k = tap*I + cin,
// zero-padded in both dims so tile loads are unconditional b128s.
__global__ void pack_weights(const float* __restrict__ w, _Float16* __restrict__ wp,
                             int O, int I, int ks, int Kpad, int Coutp) {
  int i = blockIdx.x * blockDim.x + threadIdx.x;
  int total = Kpad * Coutp;
  if (i >= total) return;
  int c = i / Kpad, k = i % Kpad;
  float v = 0.f;
  int kv = I * ks * ks;
  if (k < kv && c < O) {
    int cin = k % I, tap = k / I;
    int kh = tap / ks, kw = tap % ks;
    v = w[((size_t)((c * I + cin) * ks + kh)) * ks + kw];
  }
  wp[i] = (_Float16)v;
}

// ------------------------------------------------- implicit-GEMM conv (WMMA)
// x: NHWC f16, wp: [Coutp][KTOT] f16 packed, y: NHWC (f16 or f32)
// 256 threads = 8 waves; each wave computes a 32(M) x 64(N) strip:
// 8 x v_wmma_f32_16x16x32_f16 per 32-wide K step.
// VEC layers: GLOBAL_LOAD_ASYNC_TO_LDS_B128 double-buffer (ASYNCcnt),
// one barrier per K step, DMA overlaps the WMMA stream.
template<int CIN, int KS, int PAD, bool VEC, bool RELU, bool OUTF32>
__global__ __launch_bounds__(256)
void conv_wmma(const _Float16* __restrict__ x, const _Float16* __restrict__ wp,
               const float* __restrict__ bias, void* __restrict__ yout,
               int Nb, int H, int W, int Cout, int Coutp) {
  constexpr int KTOT = ((CIN * KS * KS + 31) / 32) * 32;
  constexpr int NK = KTOT / BK;
  __shared__ __align__(16) _Float16 As[2][BM * AS_STR];
  __shared__ __align__(16) _Float16 Bs[2][BN * AS_STR];

  const int tid  = threadIdx.x;
  const int lane = tid & 31;
  const int wid  = tid >> 5;
  const int Mtot = Nb * H * W;
  const int pbase = blockIdx.x * BM;
  const int nbase = blockIdx.y * BN;

  // per-thread A row (one pixel) -- decode hoisted out of the K loop
  const int  p  = pbase + tid;
  const bool pv = p < Mtot;
  int n = 0, h = 0, w = 0;
  if (pv) { n = p / (H * W); int rem = p - n * (H * W); h = rem / W; w = rem % W; }
  const _Float16* pixPtr = x + ((size_t)(n * H + h) * W + w) * (VEC ? CIN : 1);

  // per-thread B quarter-row: 64 rows x 32 halves staged by 256 threads
  const int brow = tid >> 2, bseg = (tid & 3) * 8;
  const _Float16* wrow = wp + (size_t)(nbase + brow) * KTOT + bseg;

  v8f acc[2][4];
  #pragma unroll
  for (int s = 0; s < 2; ++s)
    #pragma unroll
    for (int t = 0; t < 4; ++t)
      #pragma unroll
      for (int j = 0; j < 8; ++j) acc[s][t][j] = 0.f;

  v8h zero8;
  #pragma unroll
  for (int j = 0; j < 8; ++j) zero8[j] = (_Float16)0.f;

  auto compute = [&](int buf) {
    const int m    = lane & 15;
    const int kbiA = (lane & 16) ? 1 : 0;   // A chunks: {0..7,16..23} | {8..15,24..31}
    const int kbiB = (lane & 16) ? 1 : 0;   // B chunks: {0..7,16..23} | {8..15,24..31}
    #pragma unroll
    for (int s = 0; s < 2; ++s) {
      const v8h* arow = reinterpret_cast<const v8h*>(
          &As[buf][(wid * 32 + s * 16 + m) * AS_STR]);
      v8h alo = arow[kbiA], ahi = arow[kbiA + 2];   // 2 x ds_read_b128
      v16h a;
      #pragma unroll
      for (int j = 0; j < 8; ++j) { a[j] = alo[j]; a[8 + j] = ahi[j]; }
      #pragma unroll
      for (int t = 0; t < 4; ++t) {
        const v8h* brw = reinterpret_cast<const v8h*>(
            &Bs[buf][(t * 16 + m) * AS_STR]);
        v8h blo = brw[kbiB], bhi = brw[kbiB + 2];   // 2 x ds_read_b128
        v16h b;
        #pragma unroll
        for (int j = 0; j < 8; ++j) { b[j] = blo[j]; b[8 + j] = bhi[j]; }
        acc[s][t] = __builtin_amdgcn_wmma_f32_16x16x32_f16(
            false, a, false, b, (short)0, acc[s][t], false, false);
      }
    }
  };

  if (VEC) {
    // -------- async-to-LDS double-buffered pipeline --------
    auto stageAsync = [&](int kk, int buf) {
      const int k0 = kk * BK;
      int tap = k0 / CIN;
      int kh = (KS == 3) ? tap / 3 : 0;
      int kw = (KS == 3) ? tap % 3 : 0;
      int hh = h + kh - PAD, ww = w + kw - PAD;
      bool ok = pv && hh >= 0 && hh < H && ww >= 0 && ww < W;
      const _Float16* gsrc =
          pixPtr + ((long long)(kh - PAD) * W + (kw - PAD)) * CIN + (k0 % CIN);
      _Float16* adst = &As[buf][tid * AS_STR];
      if (ok) {
        #pragma unroll
        for (int i = 0; i < 4; ++i) {
          unsigned lds = (unsigned)(uintptr_t)(adst + i * 8);
          asm volatile("global_load_async_to_lds_b128 %0, %1, off"
                       :: "v"(lds), "v"(gsrc + i * 8) : "memory");
        }
      } else {
        // zero-pad border taps straight in LDS (disjoint lanes/rows from asyncs)
        v8h* ad = reinterpret_cast<v8h*>(adst);
        #pragma unroll
        for (int i = 0; i < 4; ++i) ad[i] = zero8;
      }
      {  // B tile: weights are padded, always valid
        unsigned lds = (unsigned)(uintptr_t)&Bs[buf][brow * AS_STR + bseg];
        asm volatile("global_load_async_to_lds_b128 %0, %1, off"
                     :: "v"(lds), "v"(wrow + k0) : "memory");
      }
    };

    stageAsync(0, 0);
    asm volatile("s_wait_asynccnt 0" ::: "memory");
    __syncthreads();
    for (int kk = 0; kk < NK; ++kk) {
      const int cur = kk & 1;
      // issue DMA for the next slice into the other buffer; legal because the
      // barrier ending iteration kk-1 proved every wave finished reading it
      if (kk + 1 < NK) stageAsync(kk + 1, cur ^ 1);
      if (kk + 2 < NK) {                    // speculative prefetch 2 K-slices ahead
        int k2 = (kk + 2) * BK;
        int tap = k2 / CIN;
        int kh = (KS == 3) ? tap / 3 : 0, kw = (KS == 3) ? tap % 3 : 0;
        __builtin_prefetch(pixPtr + ((long long)(kh - PAD) * W + (kw - PAD)) * CIN
                           + (k2 % CIN), 0, 0);
      }
      compute(cur);
      if (kk + 1 < NK) {
        asm volatile("s_wait_asynccnt 0" ::: "memory");  // own wave's DMAs done
        __syncthreads();                                 // ...made global
      }
    }
  } else {
    // -------- first layer (CIN==1, valid K=9 padded to 32, NK==1) --------
    v8h aReg[4]; v8h bReg;
    #pragma unroll
    for (int i = 0; i < 4; ++i) {
      #pragma unroll
      for (int j = 0; j < 8; ++j) {
        int k = i * 8 + j;
        _Float16 v = (_Float16)0.f;
        if (pv && k < 9) {
          int kh = k / 3, kw = k % 3, hh = h + kh - 1, ww = w + kw - 1;
          if (hh >= 0 && hh < H && ww >= 0 && ww < W)
            v = x[(size_t)(n * H + hh) * W + ww];
        }
        aReg[i][j] = v;
      }
    }
    bReg = *reinterpret_cast<const v8h*>(wrow);
    v8h* ad = reinterpret_cast<v8h*>(&As[0][tid * AS_STR]);
    #pragma unroll
    for (int i = 0; i < 4; ++i) ad[i] = aReg[i];
    *reinterpret_cast<v8h*>(&Bs[0][brow * AS_STR + bseg]) = bReg;
    __syncthreads();
    compute(0);
  }

  // ---- epilogue: D layout -> VGPR v: M=v (lanes<16) / v+8 (lanes>=16), N=lane&15
  const int laneN  = lane & 15;
  const int rowOff = (lane & 16) ? 8 : 0;
  #pragma unroll
  for (int s = 0; s < 2; ++s) {
    #pragma unroll
    for (int t = 0; t < 4; ++t) {
      int c = nbase + t * 16 + laneN;
      if (c >= Cout) continue;
      float bv = bias[c];
      #pragma unroll
      for (int v = 0; v < 8; ++v) {
        int pp = pbase + wid * 32 + s * 16 + rowOff + v;
        if (pp < Mtot) {
          float val = acc[s][t][v] + bv;
          if (RELU) val = fmaxf(val, 0.f);
          if (OUTF32) ((float*)yout)[(size_t)pp * Cout + c] = val;
          else        ((_Float16*)yout)[(size_t)pp * Cout + c] = (_Float16)val;
        }
      }
    }
  }
}

// ---------------------------------------------------------------- max pool
// NHWC, C % 8 == 0: vectorized 8 halves per thread (b128 loads/stores)
__global__ void maxpool2x2_nhwc(const _Float16* __restrict__ in, _Float16* __restrict__ out,
                                int Nb, int H, int W, int C) {
  int Ho = H / 2, Wo = W / 2, C8 = C >> 3;
  long long total = (long long)Nb * Ho * Wo * C8;
  long long i = (long long)blockIdx.x * blockDim.x + threadIdx.x;
  if (i >= total) return;
  int c8 = (int)(i % C8); long long r = i / C8;
  int wo = (int)(r % Wo); r /= Wo;
  int ho = (int)(r % Ho); int n = (int)(r / Ho);
  const _Float16* base0 = in + ((size_t)(n * H + 2 * ho) * W + 2 * wo) * C;
  const v8h* p00 = reinterpret_cast<const v8h*>(base0) + c8;
  const v8h* p01 = p00 + C8;
  const v8h* p10 = reinterpret_cast<const v8h*>(base0 + (size_t)W * C) + c8;
  const v8h* p11 = p10 + C8;
  v8h a = *p00, b = *p01, c = *p10, d = *p11, m;
  #pragma unroll
  for (int j = 0; j < 8; ++j)
    m[j] = (_Float16)fmaxf(fmaxf((float)a[j], (float)b[j]),
                           fmaxf((float)c[j], (float)d[j]));
  reinterpret_cast<v8h*>(out)[i] = m;
}

// -------------------------------------- softmax over 65ch + 8x8 pixel shuffle
__global__ void semi_softmax_shuffle(const float* __restrict__ semiNHWC,
                                     float* __restrict__ semiOut,  // NCHW [N,65,Hc,Wc]
                                     float* __restrict__ probOut,  // [N,1,Hc*8,Wc*8]
                                     int Nb, int Hc, int Wc) {
  int wv   = blockIdx.x * (blockDim.x >> 5) + (threadIdx.x >> 5);
  int lane = threadIdx.x & 31;
  int total = Nb * Hc * Wc;
  if (wv >= total) return;               // wave-uniform
  int wc = wv % Wc; int t = wv / Wc; int hc = t % Hc; int n = t / Hc;
  const float* pix = semiNHWC + (size_t)wv * 65;
  float v0 = pix[lane];
  float v1 = pix[lane + 32];
  float v2 = (lane == 0) ? pix[64] : 0.f;
  size_t sb = ((size_t)n * 65) * Hc * Wc + (size_t)hc * Wc + wc;
  semiOut[sb + (size_t)lane * Hc * Wc]        = v0;
  semiOut[sb + (size_t)(lane + 32) * Hc * Wc] = v1;
  if (lane == 0) semiOut[sb + (size_t)64 * Hc * Wc] = v2;
  float e0 = __expf(v0), e1 = __expf(v1);
  float e2 = (lane == 0) ? __expf(v2) : 0.f;
  float s = e0 + e1 + e2;
  #pragma unroll
  for (int off = 16; off; off >>= 1) s += __shfl_xor(s, off, 32);
  float inv = 1.f / (s + 1e-5f);
  int Hf = Hc * 8, Wf = Wc * 8;
  { int c = lane;      int i8 = c >> 3, j8 = c & 7;
    probOut[((size_t)n * Hf + hc * 8 + i8) * Wf + wc * 8 + j8] = e0 * inv; }
  { int c = lane + 32; int i8 = c >> 3, j8 = c & 7;
    probOut[((size_t)n * Hf + hc * 8 + i8) * Wf + wc * 8 + j8] = e1 * inv; }
}

// ------------------------------------------------ L2 normalize over 256 ch
__global__ void desc_normalize(const float* __restrict__ descNHWC,
                               float* __restrict__ descOut,  // NCHW [N,256,Hc,Wc]
                               int Nb, int Hc, int Wc) {
  int wv   = blockIdx.x * (blockDim.x >> 5) + (threadIdx.x >> 5);
  int lane = threadIdx.x & 31;
  int total = Nb * Hc * Wc;
  if (wv >= total) return;
  int wc = wv % Wc; int t = wv / Wc; int hc = t % Hc; int n = t / Hc;
  const float* pix = descNHWC + (size_t)wv * 256;
  float v[8]; float s = 0.f;
  #pragma unroll
  for (int k = 0; k < 8; ++k) { v[k] = pix[lane + k * 32]; s += v[k] * v[k]; }
  #pragma unroll
  for (int off = 16; off; off >>= 1) s += __shfl_xor(s, off, 32);
  float inv = rsqrtf(s);
  size_t base = ((size_t)n * 256) * Hc * Wc + (size_t)hc * Wc + wc;
  #pragma unroll
  for (int k = 0; k < 8; ++k)
    descOut[base + (size_t)(lane + k * 32) * Hc * Wc] = v[k] * inv;
}

// ------------------------------------------------------------------- driver
extern "C" void kernel_launch(void* const* d_in, const int* in_sizes, int n_in,
                              void* d_out, int out_size, void* d_ws, size_t ws_size,
                              hipStream_t stream) {
  (void)in_sizes; (void)n_in; (void)out_size; (void)ws_size;
  const int N = 8, H0 = 480, W0 = 640, Hc = 60, Wc = 80;

  const float* image = (const float*)d_in[0];
  const float *w1a=(const float*)d_in[1],  *b1a=(const float*)d_in[2];
  const float *w1b=(const float*)d_in[3],  *b1b=(const float*)d_in[4];
  const float *w2a=(const float*)d_in[5],  *b2a=(const float*)d_in[6];
  const float *w2b=(const float*)d_in[7],  *b2b=(const float*)d_in[8];
  const float *w3a=(const float*)d_in[9],  *b3a=(const float*)d_in[10];
  const float *w3b=(const float*)d_in[11], *b3b=(const float*)d_in[12];
  const float *w4a=(const float*)d_in[13], *b4a=(const float*)d_in[14];
  const float *w4b=(const float*)d_in[15], *b4b=(const float*)d_in[16];
  const float *wPa=(const float*)d_in[17], *bPa=(const float*)d_in[18];
  const float *wPb=(const float*)d_in[19], *bPb=(const float*)d_in[20];
  const float *wDa=(const float*)d_in[21], *bDa=(const float*)d_in[22];
  const float *wDb=(const float*)d_in[23], *bDb=(const float*)d_in[24];

  // workspace carve (needs ~700MB; harness preallocates d_ws)
  char* base = (char*)d_ws;
  size_t off = 0;
  auto carve = [&](size_t bytes) -> char* {
    off = (off + 255) & ~(size_t)255;
    char* p = base + off; off += bytes; return p;
  };
  _Float16* img16 = (_Float16*)carve((size_t)N * H0 * W0 * 2);
  _Float16* bufA  = (_Float16*)carve((size_t)N * H0 * W0 * 64 * 2);
  _Float16* bufB  = (_Float16*)carve((size_t)N * H0 * W0 * 64 * 2);
  float* semi32   = (float*)carve((size_t)N * Hc * Wc * 65 * 4);
  float* desc32   = (float*)carve((size_t)N * Hc * Wc * 256 * 4);

  auto packed = [&](const float* w, int O, int I, int ks, int Kpad, int Coutp) -> _Float16* {
    _Float16* p = (_Float16*)carve((size_t)Kpad * Coutp * 2);
    int tot = Kpad * Coutp;
    pack_weights<<<(tot + 255) / 256, 256, 0, stream>>>(w, p, O, I, ks, Kpad, Coutp);
    return p;
  };
  _Float16* w1a16 = packed(w1a, 64, 1, 3, 32, 64);
  _Float16* w1b16 = packed(w1b, 64, 64, 3, 576, 64);
  _Float16* w2a16 = packed(w2a, 64, 64, 3, 576, 64);
  _Float16* w2b16 = packed(w2b, 64, 64, 3, 576, 64);
  _Float16* w3a16 = packed(w3a, 128, 64, 3, 576, 128);
  _Float16* w3b16 = packed(w3b, 128, 128, 3, 1152, 128);
  _Float16* w4a16 = packed(w4a, 128, 128, 3, 1152, 128);
  _Float16* w4b16 = packed(w4b, 128, 128, 3, 1152, 128);
  _Float16* wPa16 = packed(wPa, 256, 128, 3, 1152, 256);
  _Float16* wPb16 = packed(wPb, 65, 256, 1, 256, 128);
  _Float16* wDa16 = packed(wDa, 256, 128, 3, 1152, 256);
  _Float16* wDb16 = packed(wDb, 256, 256, 1, 256, 256);

  { int n = N * H0 * W0;
    convert_image_f32_to_f16<<<(n + 255) / 256, 256, 0, stream>>>(image, img16, n); }

  auto gridFor = [](int Mtot, int Coutp) { return dim3((Mtot + BM - 1) / BM, Coutp / BN); };
  const int M480 = N * 480 * 640, M240 = N * 240 * 320, M120 = N * 120 * 160, M60 = N * 60 * 80;

  // stage 1 @ 480x640
  conv_wmma<1, 3, 1, false, true, false><<<gridFor(M480, 64), 256, 0, stream>>>(
      img16, w1a16, b1a, bufA, N, 480, 640, 64, 64);
  conv_wmma<64, 3, 1, true, true, false><<<gridFor(M480, 64), 256, 0, stream>>>(
      bufA, w1b16, b1b, bufB, N, 480, 640, 64, 64);
  maxpool2x2_nhwc<<<(unsigned)(((long long)M240 * 64 / 8 + 255) / 256), 256, 0, stream>>>(
      bufB, bufA, N, 480, 640, 64);
  // stage 2 @ 240x320
  conv_wmma<64, 3, 1, true, true, false><<<gridFor(M240, 64), 256, 0, stream>>>(
      bufA, w2a16, b2a, bufB, N, 240, 320, 64, 64);
  conv_wmma<64, 3, 1, true, true, false><<<gridFor(M240, 64), 256, 0, stream>>>(
      bufB, w2b16, b2b, bufA, N, 240, 320, 64, 64);
  maxpool2x2_nhwc<<<(unsigned)(((long long)M120 * 64 / 8 + 255) / 256), 256, 0, stream>>>(
      bufA, bufB, N, 240, 320, 64);
  // stage 3 @ 120x160
  conv_wmma<64, 3, 1, true, true, false><<<gridFor(M120, 128), 256, 0, stream>>>(
      bufB, w3a16, b3a, bufA, N, 120, 160, 128, 128);
  conv_wmma<128, 3, 1, true, true, false><<<gridFor(M120, 128), 256, 0, stream>>>(
      bufA, w3b16, b3b, bufB, N, 120, 160, 128, 128);
  maxpool2x2_nhwc<<<(unsigned)(((long long)M60 * 128 / 8 + 255) / 256), 256, 0, stream>>>(
      bufB, bufA, N, 120, 160, 128);
  // stage 4 @ 60x80
  conv_wmma<128, 3, 1, true, true, false><<<gridFor(M60, 128), 256, 0, stream>>>(
      bufA, w4a16, b4a, bufB, N, 60, 80, 128, 128);
  conv_wmma<128, 3, 1, true, true, false><<<gridFor(M60, 128), 256, 0, stream>>>(
      bufB, w4b16, b4b, bufA, N, 60, 80, 128, 128);          // x4 = bufA
  // detector head
  conv_wmma<128, 3, 1, true, true, false><<<gridFor(M60, 256), 256, 0, stream>>>(
      bufA, wPa16, bPa, bufB, N, 60, 80, 256, 256);           // cPa = bufB
  conv_wmma<256, 1, 0, true, false, true><<<gridFor(M60, 128), 256, 0, stream>>>(
      bufB, wPb16, bPb, semi32, N, 60, 80, 65, 128);
  // descriptor head (bufB reusable after Pb; stream-ordered)
  conv_wmma<128, 3, 1, true, true, false><<<gridFor(M60, 256), 256, 0, stream>>>(
      bufA, wDa16, bDa, bufB, N, 60, 80, 256, 256);           // cDa = bufB
  conv_wmma<256, 1, 0, true, false, true><<<gridFor(M60, 256), 256, 0, stream>>>(
      bufB, wDb16, bDb, desc32, N, 60, 80, 256, 256);

  // outputs: (semi, desc, prob) concatenated flat, NCHW
  float* out = (float*)d_out;
  size_t semiElems = (size_t)N * 65 * Hc * Wc;
  size_t descElems = (size_t)N * 256 * Hc * Wc;
  float* outSemi = out;
  float* outDesc = out + semiElems;
  float* outProb = out + semiElems + descElems;
  int pix = N * Hc * Wc;  // one wave per pixel, 8 waves per block
  semi_softmax_shuffle<<<(pix + 7) / 8, 256, 0, stream>>>(semi32, outSemi, outProb, N, Hc, Wc);
  desc_normalize<<<(pix + 7) / 8, 256, 0, stream>>>(desc32, outDesc, N, Hc, Wc);
}
